// HyperAttentionHead_44693429682284
// MI455X (gfx1250) — compile-verified, matrix operand order checked by reading, not run on hardware
//
#include <hip/hip_runtime.h>
#include <hip/hip_bf16.h>
#include <math.h>

// ---------------- problem constants ----------------
static constexpr int B_  = 8;
static constexpr int N_  = 1024;   // tokens
static constexpr int M_  = 1024;   // queries
static constexpr int H_  = 8;
static constexpr int NB_ = 12;
static constexpr int R_  = 24;
static constexpr int DV_ = 1280;
static constexpr int DH_ = 64;
static constexpr int DD_ = 128;
static constexpr int CT_ = 256;    // ctx dim
static constexpr int KR_ = NB_ * R_;   // 288
static constexpr int ZC_ = 2 * KR_;    // 576 (K-path | V-path)

// ---------------- workspace layout (bytes) ----------------
static constexpr size_t OFF_AK  = 0;
static constexpr size_t OFF_AV  = 4096;
static constexpr size_t OFF_XBF = 8192;                                   // B*N*DV bf16  = 20.97 MB
static constexpr size_t OFF_QBF = OFF_XBF + (size_t)B_*N_*DV_*2;          // B*M*H*DH bf16 = 8.39 MB
static constexpr size_t OFF_WT  = OFF_QBF + (size_t)B_*M_*H_*DH_*2;       // ZC*DV bf16    = 1.47 MB
static constexpr size_t OFF_GTK = OFF_WT  + (size_t)ZC_*DV_*2;            // B*H*DH*KR bf16
static constexpr size_t OFF_GTV = OFF_GTK + (size_t)B_*H_*DH_*KR_*2;
static constexpr size_t OFF_Z   = OFF_GTV + (size_t)B_*H_*DH_*KR_*2;      // B*N*ZC bf16
static constexpr size_t OFF_K   = OFF_Z   + (size_t)B_*N_*ZC_*2;          // [B,H,N,DH] bf16
static constexpr size_t OFF_VT  = OFF_K   + (size_t)B_*H_*N_*DH_*2;       // [B,H,DH,N] bf16

// ---------------- types / helpers ----------------
typedef __bf16 bf16_t;
typedef bf16_t v16bf __attribute__((ext_vector_type(16)));
typedef bf16_t v8bf  __attribute__((ext_vector_type(8)));
typedef float  v8f   __attribute__((ext_vector_type(8)));
typedef int    v4i_vs __attribute__((__vector_size__(16)));

#if defined(__has_builtin)
#  if __has_builtin(__builtin_amdgcn_global_load_async_to_lds_b128)
#    define HAVE_ASYNC_LDS 1
#  endif
#endif
#ifndef HAVE_ASYNC_LDS
#  define HAVE_ASYNC_LDS 0
#endif

__device__ __forceinline__ v8bf ld8(const bf16_t* p) { return *(const v8bf*)p; }

__device__ __forceinline__ v16bf mk16(v8bf lo, v8bf hi) {
  v16bf r;
#pragma unroll
  for (int i = 0; i < 8; ++i) { r[i] = lo[i]; r[i + 8] = hi[i]; }
  return r;
}

__device__ __forceinline__ v8f z8() {
  v8f r;
#pragma unroll
  for (int i = 0; i < 8; ++i) r[i] = 0.f;
  return r;
}

__device__ __forceinline__ v8f wmma_bf16(v16bf a, v16bf b, v8f c) {
  return __builtin_amdgcn_wmma_f32_16x16x32_bf16(false, a, false, b, (short)0, c, false, false);
}

__device__ __forceinline__ float gelu_exact(float x) {
  return 0.5f * x * (1.f + erff(x * 0.70710678118654752f));
}

// 16-byte global -> LDS copy (async on gfx1250 when available)
__device__ __forceinline__ void cp16_g2l(const bf16_t* g, bf16_t* l) {
#if HAVE_ASYNC_LDS
  __builtin_amdgcn_global_load_async_to_lds_b128(
      (__attribute__((address_space(1))) v4i_vs*)g,
      (__attribute__((address_space(3))) v4i_vs*)l, 0, 0);
#else
  *(v8bf*)l = *(const v8bf*)g;
#endif
}

__device__ __forceinline__ void wait_async_le2() {
#if HAVE_ASYNC_LDS
#  if __has_builtin(__builtin_amdgcn_s_wait_asynccnt)
  __builtin_amdgcn_s_wait_asynccnt(2);
#  else
  asm volatile("s_wait_asynccnt 0x2" ::: "memory");
#  endif
#endif
}

__device__ __forceinline__ void wait_async_le0() {
#if HAVE_ASYNC_LDS
#  if __has_builtin(__builtin_amdgcn_s_wait_asynccnt)
  __builtin_amdgcn_s_wait_asynccnt(0);
#  else
  asm volatile("s_wait_asynccnt 0x0" ::: "memory");
#  endif
#endif
}

// ---------------- small kernels ----------------
__global__ void cvt_f2bf_kernel(const float* __restrict__ src, bf16_t* __restrict__ dst, int n) {
  for (int i = blockIdx.x * blockDim.x + threadIdx.x; i < n; i += gridDim.x * blockDim.x)
    dst[i] = (bf16_t)src[i];
}

// Wt[c][d], c in [0,576): cols 0..287 from VK, 288..575 from VV. Wt[c][d] = V?[k][d][r]
__global__ void build_wt_kernel(const float* __restrict__ VK, const float* __restrict__ VV,
                                bf16_t* __restrict__ Wt) {
  const int n = ZC_ * DV_;
  for (int idx = blockIdx.x * blockDim.x + threadIdx.x; idx < n; idx += gridDim.x * blockDim.x) {
    int c = idx / DV_, d = idx - DV_ * c;
    int c2 = (c < KR_) ? c : c - KR_;
    int k = c2 / R_, r = c2 - R_ * k;
    const float* src = (c < KR_) ? VK : VV;
    Wt[idx] = (bf16_t)src[((size_t)k * DV_ + d) * R_ + r];
  }
}

__device__ __forceinline__ float block_sum256(float v, float* sred, int t) {
  sred[t] = v;
  __syncthreads();
#pragma unroll
  for (int s = 128; s > 0; s >>= 1) {
    if (t < s) sred[t] += sred[t + s];
    __syncthreads();
  }
  float r = sred[0];
  __syncthreads();
  return r;
}

// one block per batch, 256 threads: MLP + LN + alpha softmaxes
__global__ void ctx_alpha_kernel(const float* __restrict__ depth,
                                 const float* __restrict__ w1, const float* __restrict__ b1,
                                 const float* __restrict__ g1, const float* __restrict__ be1,
                                 const float* __restrict__ w2, const float* __restrict__ b2,
                                 const float* __restrict__ g2, const float* __restrict__ be2,
                                 const float* __restrict__ wck, const float* __restrict__ bck,
                                 const float* __restrict__ wcv, const float* __restrict__ bcv,
                                 float* __restrict__ aK, float* __restrict__ aV) {
  __shared__ float sA[256], sB[256], sred[256];
  const int t = threadIdx.x, b = blockIdx.x;
  if (t < DD_) sA[t] = depth[b * DD_ + t];
  __syncthreads();
  float h1 = 0.f;
  if (t < DD_) {
    float acc = b1[t];
    for (int i = 0; i < DD_; ++i) acc += sA[i] * w1[i * DD_ + t];
    h1 = gelu_exact(acc);
  }
  float s  = block_sum256(t < DD_ ? h1 : 0.f, sred, t);
  float mu = s / (float)DD_;
  float d  = (t < DD_) ? h1 - mu : 0.f;
  float vs = block_sum256(d * d, sred, t);
  float inv = rsqrtf(vs / (float)DD_ + 1e-5f);
  if (t < DD_) sB[t] = d * inv * g1[t] + be1[t];
  __syncthreads();
  float acc = b2[t];
  for (int i = 0; i < DD_; ++i) acc += sB[i] * w2[i * CT_ + t];
  float h2 = gelu_exact(acc);
  s  = block_sum256(h2, sred, t);
  mu = s / (float)CT_;
  d  = h2 - mu;
  vs = block_sum256(d * d, sred, t);
  inv = rsqrtf(vs / (float)CT_ + 1e-5f);
  sA[t] = d * inv * g2[t] + be2[t];  // ctx
  __syncthreads();
  // alpha_K
  if (t < H_ * NB_) {
    float a = bck[t];
    for (int i = 0; i < CT_; ++i) a += sA[i] * wck[i * (H_ * NB_) + t];
    sB[t] = a;
  }
  __syncthreads();
  if (t < H_) {
    float mx = -1e30f;
    for (int k = 0; k < NB_; ++k) mx = fmaxf(mx, sB[t * NB_ + k]);
    float sm = 0.f, e[NB_];
    for (int k = 0; k < NB_; ++k) { e[k] = __expf(sB[t * NB_ + k] - mx); sm += e[k]; }
    for (int k = 0; k < NB_; ++k) aK[(b * H_ + t) * NB_ + k] = e[k] / sm;
  }
  __syncthreads();
  // alpha_V
  if (t < H_ * NB_) {
    float a = bcv[t];
    for (int i = 0; i < CT_; ++i) a += sA[i] * wcv[i * (H_ * NB_) + t];
    sB[t] = a;
  }
  __syncthreads();
  if (t < H_) {
    float mx = -1e30f;
    for (int k = 0; k < NB_; ++k) mx = fmaxf(mx, sB[t * NB_ + k]);
    float sm = 0.f, e[NB_];
    for (int k = 0; k < NB_; ++k) { e[k] = __expf(sB[t * NB_ + k] - mx); sm += e[k]; }
    for (int k = 0; k < NB_; ++k) aV[(b * H_ + t) * NB_ + k] = e[k] / sm;
  }
}

// Gt[b,h][f][k*24+r] = alpha[b,h,k] * U[k,f,r]   (stored f-major so GEMM K-dim is contiguous)
__global__ void build_g_kernel(const float* __restrict__ aK, const float* __restrict__ aV,
                               const float* __restrict__ UK, const float* __restrict__ UV,
                               bf16_t* __restrict__ GtK, bf16_t* __restrict__ GtV) {
  const int bh = blockIdx.x;
  const int b = bh >> 3, h = bh & 7;
  const float* alK = aK + (b * H_ + h) * NB_;
  const float* alV = aV + (b * H_ + h) * NB_;
  const size_t base = (size_t)bh * DH_ * KR_;
  for (int idx = threadIdx.x; idx < DH_ * KR_; idx += blockDim.x) {
    int f = idx / KR_, c = idx - KR_ * f;
    int k = c / R_, r = c - R_ * k;
    GtK[base + idx] = (bf16_t)(alK[k] * UK[((size_t)k * DH_ + f) * R_ + r]);
    GtV[base + idx] = (bf16_t)(alV[k] * UV[((size_t)k * DH_ + f) * R_ + r]);
  }
}

// ---------------- Z = Xbf @ Wt^T : [B,1024,1280] x [576,1280]^T -> Z[B,1024,576] ----------------
// grid: B * (1024/128) * (576/64) = 576 blocks, 256 thr (8 waves), wave = 16 rows x 64 cols
__global__ void zgemm_kernel(const bf16_t* __restrict__ Xbf, const bf16_t* __restrict__ Wt,
                             bf16_t* __restrict__ Z) {
  const int gid = blockIdx.x;
  const int b = gid / 72, rem = gid - 72 * b;
  const int mb = rem / 9, cb = rem - 9 * mb;
  const int lane = threadIdx.x & 31, wv = threadIdx.x >> 5;
  const int lh = lane >> 4, ln = lane & 15;
  const int mbase = mb * 128 + wv * 16, cbase = cb * 64;

  v8f acc[4];
#pragma unroll
  for (int t = 0; t < 4; ++t) acc[t] = z8();

  const bf16_t* Arow = Xbf + ((size_t)b * N_ + mbase + ln) * DV_ + lh * 8;
  for (int d = 0; d < DV_; d += 32) {
    __builtin_prefetch(Arow + d + 256, 0, 1);
    v16bf a = mk16(ld8(Arow + d), ld8(Arow + d + 16));
#pragma unroll
    for (int t = 0; t < 4; ++t) {
      const bf16_t* Bp = Wt + (size_t)(cbase + 16 * t + ln) * DV_ + d + lh * 8;
      v16bf bb = mk16(ld8(Bp), ld8(Bp + 16));
      acc[t] = wmma_bf16(a, bb, acc[t]);
    }
  }
#pragma unroll
  for (int t = 0; t < 4; ++t)
#pragma unroll
    for (int v = 0; v < 8; ++v) {
      int row = v + 8 * lh;
      Z[((size_t)b * N_ + mbase + row) * ZC_ + cbase + 16 * t + ln] = (bf16_t)acc[t][v];
    }
}

// ---------------- K/V = Z @ Gt^T: per (b,h) [1024,288]x[288,64] ----------------
// grid: B*H*(1024/16)=4096 blocks, 64 thr (wave0 -> K, wave1 -> V^T)
__global__ void kvgemm_kernel(const bf16_t* __restrict__ Z, const bf16_t* __restrict__ GtK,
                              const bf16_t* __restrict__ GtV, bf16_t* __restrict__ Kb,
                              bf16_t* __restrict__ Vt) {
  const int gid = blockIdx.x;
  const int bh = gid >> 6, mt = gid & 63, mbase = mt * 16;
  const int wv = threadIdx.x >> 5, lane = threadIdx.x & 31;
  const int lh = lane >> 4, ln = lane & 15;
  const bf16_t* Gt = ((wv == 0) ? GtK : GtV) + (size_t)bh * DH_ * KR_;
  const int coff = (wv == 0) ? 0 : KR_;
  const bf16_t* Arow = Z + ((size_t)(bh >> 3) * N_ + mbase + ln) * ZC_ + coff + lh * 8;

  v8f acc[4];
#pragma unroll
  for (int t = 0; t < 4; ++t) acc[t] = z8();

  for (int c = 0; c < KR_; c += 32) {
    v16bf a = mk16(ld8(Arow + c), ld8(Arow + c + 16));
#pragma unroll
    for (int t = 0; t < 4; ++t) {
      const bf16_t* Bp = Gt + (size_t)(16 * t + ln) * KR_ + c + lh * 8;
      v16bf bb = mk16(ld8(Bp), ld8(Bp + 16));
      acc[t] = wmma_bf16(a, bb, acc[t]);
    }
  }
  if (wv == 0) {
#pragma unroll
    for (int t = 0; t < 4; ++t)
#pragma unroll
      for (int v = 0; v < 8; ++v) {
        int row = v + 8 * lh;
        Kb[((size_t)bh * N_ + mbase + row) * DH_ + 16 * t + ln] = (bf16_t)acc[t][v];
      }
  } else {
#pragma unroll
    for (int t = 0; t < 4; ++t)
#pragma unroll
      for (int v = 0; v < 8; ++v) {
        int row = v + 8 * lh;
        Vt[((size_t)bh * DH_ + 16 * t + ln) * N_ + mbase + row] = (bf16_t)acc[t][v];
      }
  }
}

// ---------------- flash attention ----------------
// grid: B*H*(1024/128)=512 blocks, 256 thr (8 waves). Block = one (b,h) x 128 M rows.
// K/V chunks (32 tokens) are double-buffered in LDS, shared by all 8 waves
// (8x L2 traffic reduction), streamed with async-to-LDS when available.
__global__ void attn_kernel(const bf16_t* __restrict__ Qbf, const bf16_t* __restrict__ Kb,
                            const bf16_t* __restrict__ Vt, float* __restrict__ out) {
  __shared__ __align__(16) bf16_t sK[2][32 * 64];   // [token][dim]
  __shared__ __align__(16) bf16_t sV[2][64 * 32];   // [dim][token]
  __shared__ __align__(16) bf16_t Pbuf[8][16 * 32];
  const int gid = blockIdx.x;
  const int bh = gid >> 3, mc = gid & 7;
  const int b = bh >> 3, h = bh & 7;
  const int tid = threadIdx.x;
  const int wv = tid >> 5, lane = tid & 31;
  const int lh = lane >> 4, ln = lane & 15;
  const int mbase = mc * 128 + wv * 16;

  const bf16_t* Kg = Kb + (size_t)bh * N_ * DH_;
  const bf16_t* Vg = Vt + (size_t)bh * DH_ * N_;
  // staging assignment: 256 threads x 16B each for K (32x64) and V (64x32)
  const int ktok = tid >> 3, koff = (tid & 7) * 8;
  const int vf = tid >> 2, voff = (tid & 3) * 8;

  const bf16_t* qp = Qbf + (((size_t)b * M_ + mbase + ln) * H_ + h) * DH_ + lh * 8;
  v16bf qa0 = mk16(ld8(qp), ld8(qp + 16));
  v16bf qa1 = mk16(ld8(qp + 32), ld8(qp + 48));

  v8f O[4];
#pragma unroll
  for (int t = 0; t < 4; ++t) O[t] = z8();
  float mrun[8], lrun[8];
#pragma unroll
  for (int v = 0; v < 8; ++v) { mrun[v] = -1e30f; lrun[v] = 0.f; }

  bf16_t* P = &Pbuf[wv][0];

  // prologue: stage chunk 0
  cp16_g2l(Kg + (size_t)ktok * DH_ + koff, &sK[0][ktok * 64 + koff]);
  cp16_g2l(Vg + (size_t)vf * N_ + voff, &sV[0][vf * 32 + voff]);

  const int NCH = N_ / 32;
  for (int c = 0; c < NCH; ++c) {
    const int cur = c & 1;
    if (c + 1 < NCH) {
      const int n1 = (c + 1) * 32;
      cp16_g2l(Kg + (size_t)(n1 + ktok) * DH_ + koff, &sK[cur ^ 1][ktok * 64 + koff]);
      cp16_g2l(Vg + (size_t)vf * N_ + n1 + voff, &sV[cur ^ 1][vf * 32 + voff]);
      wait_async_le2();   // chunk c landed (in-order), chunk c+1 may be in flight
    } else {
      wait_async_le0();
    }
    __syncthreads();

    const bf16_t* sKc = &sK[cur][0];
    const bf16_t* sVc = &sV[cur][0];

    v8f S0 = z8(), S1 = z8();
    {
      const bf16_t* kp = sKc + ln * 64 + lh * 8;
      S0 = wmma_bf16(qa0, mk16(ld8(kp), ld8(kp + 16)), S0);
      S0 = wmma_bf16(qa1, mk16(ld8(kp + 32), ld8(kp + 48)), S0);
      kp += 16 * 64;
      S1 = wmma_bf16(qa0, mk16(ld8(kp), ld8(kp + 16)), S1);
      S1 = wmma_bf16(qa1, mk16(ld8(kp + 32), ld8(kp + 48)), S1);
    }

    float corr[8];
#pragma unroll
    for (int v = 0; v < 8; ++v) {
      float a = S0[v] * 0.125f, c2 = S1[v] * 0.125f;
      float mx = fmaxf(a, c2);
#pragma unroll
      for (int msk = 1; msk < 16; msk <<= 1) mx = fmaxf(mx, __shfl_xor(mx, msk, 32));
      float mn = fmaxf(mrun[v], mx);
      float cr = __expf(mrun[v] - mn);
      float p0 = __expf(a - mn), p1 = __expf(c2 - mn);
      float rs = p0 + p1;
#pragma unroll
      for (int msk = 1; msk < 16; msk <<= 1) rs += __shfl_xor(rs, msk, 32);
      lrun[v] = lrun[v] * cr + rs;
      mrun[v] = mn;
      corr[v] = cr;
      S0[v] = p0;
      S1[v] = p1;
    }
#pragma unroll
    for (int t = 0; t < 4; ++t)
#pragma unroll
      for (int v = 0; v < 8; ++v) O[t][v] *= corr[v];

    // C-layout -> A-layout via per-wave LDS bounce
#pragma unroll
    for (int v = 0; v < 8; ++v) {
      int row = v + 8 * lh;
      P[row * 32 + ln] = (bf16_t)S0[v];
      P[row * 32 + ln + 16] = (bf16_t)S1[v];
    }
    const bf16_t* pr = P + ln * 32 + lh * 8;
    v16bf pa = mk16(*(const v8bf*)pr, *(const v8bf*)(pr + 16));
#pragma unroll
    for (int t = 0; t < 4; ++t) {
      const bf16_t* vp = sVc + (16 * t + ln) * 32 + lh * 8;
      v16bf vb = mk16(ld8(vp), ld8(vp + 16));
      O[t] = wmma_bf16(pa, vb, O[t]);
    }
    __syncthreads();  // all waves done with buffer `cur` before it is re-staged
  }

#pragma unroll
  for (int t = 0; t < 4; ++t)
#pragma unroll
    for (int v = 0; v < 8; ++v) {
      int row = v + 8 * lh;
      out[((size_t)b * M_ + mbase + row) * (H_ * DH_) + h * DH_ + 16 * t + ln] =
          O[t][v] / lrun[v];
    }
}

// ---------------- launch ----------------
extern "C" void kernel_launch(void* const* d_in, const int* in_sizes, int n_in,
                              void* d_out, int out_size, void* d_ws, size_t ws_size,
                              hipStream_t stream) {
  (void)in_sizes; (void)n_in; (void)out_size; (void)ws_size;
  const float* visual = (const float*)d_in[0];
  const float* depth  = (const float*)d_in[1];
  const float* Q      = (const float*)d_in[2];
  const float* w1  = (const float*)d_in[3];
  const float* b1  = (const float*)d_in[4];
  const float* g1  = (const float*)d_in[5];
  const float* be1 = (const float*)d_in[6];
  const float* w2  = (const float*)d_in[7];
  const float* b2  = (const float*)d_in[8];
  const float* g2  = (const float*)d_in[9];
  const float* be2 = (const float*)d_in[10];
  const float* wck = (const float*)d_in[11];
  const float* bck = (const float*)d_in[12];
  const float* wcv = (const float*)d_in[13];
  const float* bcv = (const float*)d_in[14];
  const float* UK  = (const float*)d_in[15];
  const float* VK  = (const float*)d_in[16];
  const float* UV  = (const float*)d_in[17];
  const float* VV  = (const float*)d_in[18];

  char* w = (char*)d_ws;
  float*  aK  = (float*)(w + OFF_AK);
  float*  aV  = (float*)(w + OFF_AV);
  bf16_t* Xbf = (bf16_t*)(w + OFF_XBF);
  bf16_t* Qbf = (bf16_t*)(w + OFF_QBF);
  bf16_t* Wt  = (bf16_t*)(w + OFF_WT);
  bf16_t* GtK = (bf16_t*)(w + OFF_GTK);
  bf16_t* GtV = (bf16_t*)(w + OFF_GTV);
  bf16_t* Z   = (bf16_t*)(w + OFF_Z);
  bf16_t* Kb  = (bf16_t*)(w + OFF_K);
  bf16_t* Vt  = (bf16_t*)(w + OFF_VT);

  cvt_f2bf_kernel<<<2048, 256, 0, stream>>>(visual, Xbf, B_ * N_ * DV_);
  cvt_f2bf_kernel<<<2048, 256, 0, stream>>>(Q, Qbf, B_ * M_ * H_ * DH_);
  build_wt_kernel<<<720, 256, 0, stream>>>(VK, VV, Wt);
  ctx_alpha_kernel<<<B_, 256, 0, stream>>>(depth, w1, b1, g1, be1, w2, b2, g2, be2,
                                           wck, bck, wcv, bcv, aK, aV);
  build_g_kernel<<<B_ * H_, 256, 0, stream>>>(aK, aV, UK, UV, GtK, GtV);
  zgemm_kernel<<<576, 256, 0, stream>>>(Xbf, Wt, Z);
  kvgemm_kernel<<<4096, 64, 0, stream>>>(Z, GtK, GtV, Kb, Vt);
  attn_kernel<<<512, 256, 0, stream>>>(Qbf, Kb, Vt, (float*)d_out);
}
